// CausalSelfAttention_28132035788874
// MI455X (gfx1250) — matrix operand order repaired and checked by reading, not compiled, not run on hardware
//
#include <hip/hip_runtime.h>
#include <hip/hip_bf16.h>

// ---------------------------------------------------------------------------
// Causal self-attention (B=8, T=1024, C=768, H=12, D=64) for gfx1250 (MI455X).
// All GEMM-shaped math runs through v_wmma_f32_16x16x32_f16 (f16 in, f32 acc).
// ---------------------------------------------------------------------------

#define BDIM   8
#define TDIM   1024
#define CDIM   768
#define HEADS  12
#define DDIM   64
#define MROWS  (BDIM * TDIM)   // 8192
#define N_QKV  (3 * CDIM)      // 2304

typedef __attribute__((ext_vector_type(16))) _Float16 v16h;
typedef __attribute__((ext_vector_type(8)))  float    v8f;

union Frag16 { v16h v; uint4 q[2]; };

// ---- WMMA fragment builders -----------------------------------------------
// A-matrix 16x32 f16 layout (per ISA 7.12.2): lane L holds row L%16;
//   halves 0..7  = K in [kb, kb+8)      (kb = 0 for lanes 0-15, 8 for 16-31)
//   halves 8..15 = K in [16+kb, 24+kb)
// => two contiguous 16B LDS loads at (+kb) and (+16+kb).
__device__ __forceinline__ v16h frag_a(const _Float16* lds, int row0, int stride, int col0) {
  const int lane = threadIdx.x & 31;
  const int r    = lane & 15;
  const int kb   = (lane & 16) ? 8 : 0;
  const _Float16* p = lds + (row0 + r) * stride + col0 + kb;
  Frag16 f;
  f.q[0] = *(const uint4*)(p);
  f.q[1] = *(const uint4*)(p + 16);
  return f.v;
}

// B-matrix 32x16 f16 layout: lane L holds column n0 + L%16;
//   halves 0..7 = K in [kb, kb+8), halves 8..15 = K in [16+kb, 24+kb)  (kb = 0 for lanes 0-15, 8 for 16-31)
// Requires the LDS tile stored K-contiguous per column (i.e. transposed,
// stride = 32 halves), so the fragment is two contiguous 16B loads.
__device__ __forceinline__ v16h frag_b(const _Float16* lds, int n0, int stride, int col0) {
  const int lane = threadIdx.x & 31;
  const int r    = lane & 15;
  const int kb   = (lane & 16) ? 8 : 0;
  const _Float16* p = lds + (n0 + r) * stride + col0 + kb;
  Frag16 f;
  f.q[0] = *(const uint4*)(p);
  f.q[1] = *(const uint4*)(p + 16);
  return f.v;
}

__device__ __forceinline__ v8f wmma16(v16h a, v16h b, v8f c) {
  return __builtin_amdgcn_wmma_f32_16x16x32_f16(false, a, false, b, (short)0, c, false, false);
}

// ---- f32 -> f16 conversion -------------------------------------------------
__global__ __launch_bounds__(256) void cvt_f32_f16(const float* __restrict__ in,
                                                   _Float16* __restrict__ out, int n) {
  for (int i = blockIdx.x * blockDim.x + threadIdx.x; i < n; i += gridDim.x * blockDim.x)
    out[i] = (_Float16)in[i];
}

// ---- shared GEMM core: 64x64 block, 4 waves x (32x32), K-step 32 -----------
// A: f16 [M,K] row-major, Bm: f16 [K,N] row-major.
// LDS: As[64][32] (row-major, K contiguous), Bst[64][32] (transposed: n-major).
__device__ __forceinline__ void gemm_core(const _Float16* __restrict__ A,
                                          const _Float16* __restrict__ Bm,
                                          int Ncols, int Kdim,
                                          int rowBlk, int colBlk,
                                          _Float16* As, _Float16* Bst,
                                          v8f (&c)[2][2]) {
  const int tid = threadIdx.x;
  const int w   = tid >> 5;
  const int wr  = (w >> 1) * 32;   // wave row offset in block
  const int wc  = (w & 1) * 32;    // wave col offset in block

  for (int a = 0; a < 2; ++a)
    for (int b = 0; b < 2; ++b)
      for (int i = 0; i < 8; ++i) c[a][b][i] = 0.0f;

  for (int kt = 0; kt < Kdim; kt += 32) {
    // A tile 64x32: thread reads 16 contiguous f16, stores contiguous.
    for (int j = 0; j < 16; ++j) {
      int e = tid * 16 + j;                 // e = r*32 + k
      int r = e >> 5, k = e & 31;
      As[e] = (_Float16)A[(size_t)(rowBlk + r) * Kdim + kt + k];
    }
    // B tile 32x64 -> transposed into Bst[n][k] (stride 32).
    for (int j = 0; j < 16; ++j) {
      int e = tid * 16 + j;                 // e = k*64 + n  (n contiguous in global)
      int k = e >> 6, n = e & 63;
      Bst[n * 32 + k] = (_Float16)Bm[(size_t)(kt + k) * Ncols + colBlk + n];
    }
    __syncthreads();

    v16h a0 = frag_a(As, wr,       32, 0);
    v16h a1 = frag_a(As, wr + 16,  32, 0);
    v16h b0 = frag_b(Bst, wc,      32, 0);
    v16h b1 = frag_b(Bst, wc + 16, 32, 0);

    c[0][0] = wmma16(a0, b0, c[0][0]);
    c[0][1] = wmma16(a0, b1, c[0][1]);
    c[1][0] = wmma16(a1, b0, c[1][0]);
    c[1][1] = wmma16(a1, b1, c[1][1]);
    __syncthreads();
  }
}

// ---- QKV projection: qkv = xh @ w_attn + b_attn, scatter to [B,H,T,D] f16 --
__global__ __launch_bounds__(128) void gemm_qkv_kernel(const _Float16* __restrict__ A,
                                                       const _Float16* __restrict__ Bm,
                                                       const float* __restrict__ bias,
                                                       _Float16* __restrict__ Qo,
                                                       _Float16* __restrict__ Ko,
                                                       _Float16* __restrict__ Vo) {
  __shared__ _Float16 As[64 * 32];
  __shared__ _Float16 Bst[64 * 32];
  v8f c[2][2];
  const int rowBlk = blockIdx.y * 64;
  const int colBlk = blockIdx.x * 64;
  gemm_core(A, Bm, N_QKV, CDIM, rowBlk, colBlk, As, Bst, c);

  const int lane = threadIdx.x & 31;
  const int w    = threadIdx.x >> 5;
  const int wr   = (w >> 1) * 32, wc = (w & 1) * 32;
  const int hi8  = (lane & 16) ? 8 : 0;
  const int cl   = lane & 15;
  for (int a = 0; a < 2; ++a)
    for (int b = 0; b < 2; ++b)
      for (int i = 0; i < 8; ++i) {
        int row = rowBlk + wr + a * 16 + i + hi8;      // 0..8191 (b*T + t)
        int col = colBlk + wc + b * 16 + cl;           // 0..2303
        float val = c[a][b][i] + bias[col];
        int sec = col / CDIM, rem = col % CDIM;
        int hh = rem >> 6, dd = rem & 63;
        int bi = row >> 10, t = row & 1023;
        _Float16* dst = (sec == 0) ? Qo : (sec == 1) ? Ko : Vo;
        dst[(((size_t)bi * HEADS + hh) * TDIM + t) * DDIM + dd] = (_Float16)val;
      }
}

// ---- Output projection: out = yh @ w_proj + b_proj (f32 out) ---------------
__global__ __launch_bounds__(128) void gemm_out_kernel(const _Float16* __restrict__ A,
                                                       const _Float16* __restrict__ Bm,
                                                       const float* __restrict__ bias,
                                                       float* __restrict__ Out) {
  __shared__ _Float16 As[64 * 32];
  __shared__ _Float16 Bst[64 * 32];
  v8f c[2][2];
  const int rowBlk = blockIdx.y * 64;
  const int colBlk = blockIdx.x * 64;
  gemm_core(A, Bm, CDIM, CDIM, rowBlk, colBlk, As, Bst, c);

  const int lane = threadIdx.x & 31;
  const int w    = threadIdx.x >> 5;
  const int wr   = (w >> 1) * 32, wc = (w & 1) * 32;
  const int hi8  = (lane & 16) ? 8 : 0;
  const int cl   = lane & 15;
  for (int a = 0; a < 2; ++a)
    for (int b = 0; b < 2; ++b)
      for (int i = 0; i < 8; ++i) {
        int row = rowBlk + wr + a * 16 + i + hi8;
        int col = colBlk + wc + b * 16 + cl;
        Out[(size_t)row * CDIM + col] = c[a][b][i] + bias[col];
      }
}

// ---- Flash attention: one block = 64 query rows of one (b,h); 4 waves ------
__global__ __launch_bounds__(128) void attn_kernel(const _Float16* __restrict__ Q,
                                                   const _Float16* __restrict__ K,
                                                   const _Float16* __restrict__ V,
                                                   _Float16* __restrict__ Y) {
  __shared__ _Float16 Qs[64 * 64];        // Q block, [row][d], stride 64
  __shared__ _Float16 Ks[32 * 64];        // K tile,  [key][d], stride 64 (K-contig for QK^T B-frag)
  __shared__ _Float16 Vst[64 * 32];       // V tile transposed, [d][key], stride 32 (key-contig for PV B-frag)
  __shared__ _Float16 Ps[4][16 * 32];     // per-wave P tile, [row][key], stride 32

  const int tid  = threadIdx.x;
  const int lane = tid & 31;
  const int w    = tid >> 5;
  const int bh   = blockIdx.y;                       // b*HEADS + h
  const int b    = bh / HEADS, h = bh % HEADS;
  const int qbase = blockIdx.x * 64;

  const _Float16* Qg = Q + (size_t)bh * TDIM * DDIM;
  const _Float16* Kg = K + (size_t)bh * TDIM * DDIM;
  const _Float16* Vg = V + (size_t)bh * TDIM * DDIM;

  // Q block: 4096 contiguous halves = 512 x 16B.
  {
    const uint4* src = (const uint4*)(Qg + (size_t)qbase * DDIM);
    uint4* dst = (uint4*)Qs;
    for (int i = tid; i < 512; i += 128) dst[i] = src[i];
  }
  __syncthreads();

  const v16h aq0 = frag_a(Qs, w * 16, 64, 0);    // Q rows, d 0..31
  const v16h aq1 = frag_a(Qs, w * 16, 64, 32);   // Q rows, d 32..63

  v8f o[4];
  float m_i[8], l_i[8];
  for (int d = 0; d < 4; ++d)
    for (int i = 0; i < 8; ++i) o[d][i] = 0.0f;
  for (int i = 0; i < 8; ++i) { m_i[i] = -1e30f; l_i[i] = 0.0f; }

  const int hi8 = (lane & 16) ? 8 : 0;
  const int cl  = lane & 15;
  const int ntiles = qbase / 32 + 2;             // causal: keys 0 .. qbase+63

  for (int kt = 0; kt < ntiles; ++kt) {
    // K tile: 2048 contiguous halves.
    {
      const uint4* src = (const uint4*)(Kg + (size_t)kt * 32 * DDIM);
      uint4* dst = (uint4*)Ks;
      for (int i = tid; i < 256; i += 128) dst[i] = src[i];
    }
    // V tile, transposed into [d][key].
    {
      const _Float16* src = Vg + (size_t)kt * 32 * DDIM;
      int e = tid * 16;
      for (int j = 0; j < 16; ++j, ++e) {
        int key = e >> 6, d = e & 63;
        Vst[d * 32 + key] = src[e];
      }
    }
    __syncthreads();

    // S = Q . K^T   (16 q-rows x 32 keys), accumulated over d in {0..31, 32..63}.
    v8f z;
    for (int i = 0; i < 8; ++i) z[i] = 0.0f;
    v8f s0 = wmma16(aq0, frag_b(Ks, 0,  64, 0),  z);
    s0     = wmma16(aq1, frag_b(Ks, 0,  64, 32), s0);
    v8f s1 = wmma16(aq0, frag_b(Ks, 16, 64, 0),  z);
    s1     = wmma16(aq1, frag_b(Ks, 16, 64, 32), s1);

    // Online softmax (row reductions across the 16 lanes of each half-wave).
    const int kcol0 = kt * 32 + cl;
    for (int i = 0; i < 8; ++i) {
      int qrow = qbase + w * 16 + i + hi8;
      float v0 = s0[i] * 0.125f; if (kcol0      > qrow) v0 = -1e30f;
      float v1 = s1[i] * 0.125f; if (kcol0 + 16 > qrow) v1 = -1e30f;
      float tm = fmaxf(v0, v1);
      for (int off = 8; off; off >>= 1) tm = fmaxf(tm, __shfl_xor(tm, off, 32));
      float mn   = fmaxf(m_i[i], tm);
      float corr = __expf(m_i[i] - mn);
      float e0   = __expf(v0 - mn);
      float e1   = __expf(v1 - mn);
      float rs   = e0 + e1;
      for (int off = 8; off; off >>= 1) rs += __shfl_xor(rs, off, 32);
      l_i[i] = l_i[i] * corr + rs;
      m_i[i] = mn;
      for (int d = 0; d < 4; ++d) o[d][i] *= corr;
      // Re-layout P through per-wave LDS for the next WMMA's A fragment.
      Ps[w][(i + hi8) * 32 + cl]      = (_Float16)e0;
      Ps[w][(i + hi8) * 32 + 16 + cl] = (_Float16)e1;
    }

    // O += P . V   (16 x 64, K = 32 keys).
    v16h ap = frag_a(Ps[w], 0, 32, 0);
    for (int d = 0; d < 4; ++d)
      o[d] = wmma16(ap, frag_b(Vst, d * 16, 32, 0), o[d]);
    __syncthreads();
  }

  // Normalize and store Y as f16 [B*T, C] with col = h*64 + d.
  for (int i = 0; i < 8; ++i) {
    float inv = 1.0f / l_i[i];
    int t = qbase + w * 16 + i + hi8;
    size_t row = (size_t)b * TDIM + t;
    for (int d = 0; d < 4; ++d) {
      float val = o[d][i] * inv;
      Y[row * CDIM + h * DDIM + d * 16 + cl] = (_Float16)val;
    }
  }
}

// ---------------------------------------------------------------------------
extern "C" void kernel_launch(void* const* d_in, const int* in_sizes, int n_in,
                              void* d_out, int out_size, void* d_ws, size_t ws_size,
                              hipStream_t stream) {
  const float* x      = (const float*)d_in[0];   // [B,T,C]
  const float* w_attn = (const float*)d_in[1];   // [C,3C]
  const float* b_attn = (const float*)d_in[2];   // [3C]
  const float* w_proj = (const float*)d_in[3];   // [C,C]
  const float* b_proj = (const float*)d_in[4];   // [C]
  float* out = (float*)d_out;

  const int nx  = MROWS * CDIM;          // 6,291,456
  const int nwa = CDIM * N_QKV;          // 1,769,472
  const int nwp = CDIM * CDIM;           //   589,824
  const int nqkv = BDIM * HEADS * TDIM * DDIM;  // 6,291,456 per tensor

  size_t off = 0;
  auto alloc = [&](size_t bytes) -> void* {
    void* p = (char*)d_ws + off;
    off += (bytes + 255) & ~(size_t)255;
    return p;
  };
  _Float16* xh  = (_Float16*)alloc((size_t)nx  * 2);
  _Float16* wah = (_Float16*)alloc((size_t)nwa * 2);
  _Float16* wph = (_Float16*)alloc((size_t)nwp * 2);
  _Float16* Qh  = (_Float16*)alloc((size_t)nqkv * 2);
  _Float16* Kh  = (_Float16*)alloc((size_t)nqkv * 2);
  _Float16* Vh  = (_Float16*)alloc((size_t)nqkv * 2);
  _Float16* Yh  = (_Float16*)alloc((size_t)nx  * 2);
  (void)ws_size; (void)in_sizes; (void)n_in; (void)out_size;

  // 1) f32 -> f16 conversions.
  cvt_f32_f16<<<4096, 256, 0, stream>>>(x,      xh,  nx);
  cvt_f32_f16<<<2048, 256, 0, stream>>>(w_attn, wah, nwa);
  cvt_f32_f16<<<1024, 256, 0, stream>>>(w_proj, wph, nwp);

  // 2) QKV projection GEMM (8192 x 2304 x 768), scatter to [B,H,T,D].
  gemm_qkv_kernel<<<dim3(N_QKV / 64, MROWS / 64), 128, 0, stream>>>(
      xh, wah, b_attn, Qh, Kh, Vh);

  // 3) Causal flash attention per (b,h), 64 query rows per block.
  attn_kernel<<<dim3(TDIM / 64, BDIM * HEADS), 128, 0, stream>>>(Qh, Kh, Vh, Yh);

  // 4) Output projection GEMM (8192 x 768 x 768), f32 out + bias.
  gemm_out_kernel<<<dim3(CDIM / 64, MROWS / 64), 128, 0, stream>>>(
      Yh, wph, b_proj, out);
}
